// VectorQuantizer_26551487824072
// MI455X (gfx1250) — compile-verified, matrix operand order checked by reading, not run on hardware
//
#include <hip/hip_runtime.h>
#include <stdint.h>

// -------- problem constants (match reference) --------
#define K_CODES   1024      // NUM_EMBEDDINGS
#define DIM       256       // EMBEDDING_DIM
#define T_LEN     2048
#define B_SZ      32
#define TTILE     128       // t-rows per workgroup (8 waves x 16 rows)
#define NTHREADS  256
#define EBSTRIDE  264       // bf16 elems per LDS tile row: 528B = 33 x 16B chunks, stride%64 banks == 4

typedef __attribute__((ext_vector_type(16))) __bf16 v16bf;
typedef __attribute__((ext_vector_type(8)))  __bf16 v8bf;
typedef __attribute__((ext_vector_type(8)))  float  v8f;

// ---------------- CDNA5 async global->LDS helpers ----------------
// GVS addressing: mem = SADDR(64) + VADDR(32, per-lane) ; LDS dest = per-lane VGPR byte address.
__device__ __forceinline__ void async_copy_b128(uint32_t ldsByteAddr, const void* sbase,
                                                uint32_t gByteOff) {
  asm volatile("global_load_async_to_lds_b128 %0, %1, %2"
               :: "v"(ldsByteAddr), "v"(gByteOff), "s"(sbase)
               : "memory");
}
__device__ __forceinline__ void wait_async0() {
  asm volatile("s_wait_asynccnt 0" ::: "memory");
}

// Kernel 1: e_norms (exact f32), split E into bf16 hi/lo planes, zero loss slot.
__global__ __launch_bounds__(256) void vq_prep(const float* __restrict__ E,
                                               float* __restrict__ enorms,
                                               __bf16* __restrict__ ehi,
                                               __bf16* __restrict__ elo,
                                               float* __restrict__ lossSlot) {
  const int k = blockIdx.x;
  const int tid = threadIdx.x;
  const float v = E[(size_t)k * DIM + tid];
  const __bf16 hi = (__bf16)v;
  ehi[(size_t)k * DIM + tid] = hi;
  elo[(size_t)k * DIM + tid] = (__bf16)(v - (float)hi);
  float s = v * v;
#pragma unroll
  for (int off = 16; off >= 1; off >>= 1) s += __shfl_xor(s, off, 32);
  __shared__ float ws[8];
  if ((tid & 31) == 0) ws[tid >> 5] = s;
  __syncthreads();
  if (tid == 0) {
    float tot = 0.f;
#pragma unroll
    for (int i = 0; i < 8; ++i) tot += ws[i];
    enorms[k] = tot;
    if (k == 0) *lossSlot = 0.0f;
  }
}

// Assemble a v16bf WMMA operand from two 16B LDS chunks (K 0..7+8h and K 16..23+8h).
__device__ __forceinline__ v16bf load_op16(const __bf16* p0, const __bf16* p1) {
  v8bf c0 = *(const v8bf*)p0;
  v8bf c1 = *(const v8bf*)p1;
  v16bf r;
#pragma unroll
  for (int i = 0; i < 8; ++i) { r[i] = c0[i]; r[i + 8] = c1[i]; }
  return r;
}

// Main kernel: split-bf16 distance GEMM (3x v_wmma_f32_16x16x32_bf16 per 32-K step),
// double-buffered async E-tile streaming, fused argmin + gather + loss + transpose-out.
__global__ __launch_bounds__(NTHREADS) void vq_main(const float* __restrict__ X,       // [B, D, T]
                                                    const __bf16* __restrict__ ehi,    // [K, D]
                                                    const __bf16* __restrict__ elo,    // [K, D]
                                                    const float* __restrict__ E,       // [K, D] f32
                                                    const float* __restrict__ enorms,  // [K]
                                                    float* __restrict__ outQ,          // [B, D, T]
                                                    float* __restrict__ outLoss,       // [1]
                                                    float* __restrict__ outIdx) {      // [B, T]
  __shared__ __align__(16) __bf16 EtHi[2][16 * EBSTRIDE];
  __shared__ __align__(16) __bf16 EtLo[2][16 * EBSTRIDE];
  __shared__ float enLDS[K_CODES];
  __shared__ int   idxArr[TTILE];
  __shared__ float redBuf[8];

  const int tid  = threadIdx.x;          // 0..255
  const int lane = tid & 31;
  const int wave = tid >> 5;             // 0..7 -> M-subtile (16 rows each)
  const int b    = blockIdx.x >> 4;      // 16 t-tiles (of 128) per batch
  const int t0   = (blockIdx.x & 15) * TTILE;

  // Async-stage one 16-code tile (bf16 hi+lo planes) into buffer `buf`.
  // 16 rows x 512B per plane = 512 x 16B chunks; each thread issues 2 per plane.
  auto stage_tile = [&](int nt, int buf) {
    const uint32_t hiBase = (uint32_t)(uintptr_t)&EtHi[buf][0];
    const uint32_t loBase = (uint32_t)(uintptr_t)&EtLo[buf][0];
    const uint32_t g0 = (uint32_t)nt * 16u * (DIM * 2u);
#pragma unroll
    for (int f = tid; f < 512; f += NTHREADS) {
      const int c = f >> 5;                                  // code row in tile
      const int q = f & 31;                                  // 16B chunk within row
      const uint32_t gOff = g0 + (uint32_t)c * (DIM * 2u) + (uint32_t)q * 16u;
      const uint32_t lOff = ((uint32_t)c * EBSTRIDE + (uint32_t)q * 8u) * 2u;
      async_copy_b128(hiBase + lOff, ehi, gOff);
      async_copy_b128(loBase + lOff, elo, gOff);
    }
  };

  // stage e-norms once (covered by first barrier)
  for (int i = tid; i < K_CODES; i += NTHREADS) enLDS[i] = enorms[i];

  // kick off tile 0 while we build the resident A operands
  stage_tile(0, 0);

  // ---- build this wave's resident A operands (16 rows x 256 K, bf16 hi+lo) ----
  // 16-bit A layout: lane half hh selects K blocks {8hh..8hh+7} and {16+8hh..23+8hh} per 32-K step.
  const int lm = lane & 15;
  const int hh = lane >> 4;              // 0 or 1
  const float* xb = X + ((size_t)b * DIM) * T_LEN + t0 + wave * 16 + lm;
  v16bf aH[8], aL[8];
#pragma unroll
  for (int jj = 0; jj < 8; ++jj) {
#pragma unroll
    for (int i = 0; i < 16; ++i) {
      const int d = 32 * jj + 8 * hh + ((i < 8) ? i : (i + 8));
      const float x = xb[(size_t)d * T_LEN];
      const __bf16 hi = (__bf16)x;
      aH[jj][i] = hi;
      aL[jj][i] = (__bf16)(x - (float)hi);
    }
  }

  float minv[8];
  int   mini[8];
#pragma unroll
  for (int r = 0; r < 8; ++r) { minv[r] = 3.4e38f; mini[r] = 0; }

  wait_async0();
  __syncthreads();   // tile 0 + enLDS visible to all waves

  // ---- 64 N-tiles of 16 codes; double-buffered async copy overlaps the WMMAs ----
  for (int nt = 0; nt < K_CODES / 16; ++nt) {
    const int cur = nt & 1;
    if (nt + 1 < K_CODES / 16) stage_tile(nt + 1, cur ^ 1);   // prev barrier freed that buffer

    // three independent accumulation chains: hi*hi, hi*lo, lo*hi
    v8f acc0 = {}, acc1 = {}, acc2 = {};
    const __bf16* bhRow = &EtHi[cur][lm * EBSTRIDE];
    const __bf16* blRow = &EtLo[cur][lm * EBSTRIDE];
#pragma unroll
    for (int jj = 0; jj < 8; ++jj) {
      const int d0 = 32 * jj + 8 * hh;
      v16bf Bh = load_op16(bhRow + d0, bhRow + d0 + 16);
      v16bf Bl = load_op16(blRow + d0, blRow + d0 + 16);
      acc0 = __builtin_amdgcn_wmma_f32_16x16x32_bf16(false, aH[jj], false, Bh,
                                                     (short)0, acc0, false, false);
      acc1 = __builtin_amdgcn_wmma_f32_16x16x32_bf16(false, aH[jj], false, Bl,
                                                     (short)0, acc1, false, false);
      acc2 = __builtin_amdgcn_wmma_f32_16x16x32_bf16(false, aL[jj], false, Bh,
                                                     (short)0, acc2, false, false);
    }

    // scores -> running argmin. Column (code) per lane = n0 + lm; row = wave*16 + 8*hh + r.
    const int   n0   = nt * 16;
    const float en   = enLDS[n0 + lm];
    const int   code = n0 + lm;
#pragma unroll
    for (int r = 0; r < 8; ++r) {
      const float dist = en - 2.0f * (acc0[r] + (acc1[r] + acc2[r]));
      if (dist < minv[r]) { minv[r] = dist; mini[r] = code; }
    }

    wait_async0();     // own async copies into the other buffer done
    __syncthreads();   // all waves done reading cur + all copies complete
  }

  // ---- reduce argmin across the 16 lanes of each half-wave (ties -> lowest index) ----
#pragma unroll
  for (int r = 0; r < 8; ++r) {
    float v = minv[r];
    int   ix = mini[r];
#pragma unroll
    for (int off = 8; off >= 1; off >>= 1) {
      const float v2 = __shfl_xor(v, off, 32);
      const int   i2 = __shfl_xor(ix, off, 32);
      if (v2 < v || (v2 == v && i2 < ix)) { v = v2; ix = i2; }
    }
    if (lm == 0) idxArr[wave * 16 + hh * 8 + r] = ix;
  }
  __syncthreads();

  // ---- phase 2: gather codebook rows (f32, L2-hot), write transposed quantized, loss ----
  float lsum = 0.0f;
  const float* xrow = X    + ((size_t)b * DIM) * T_LEN + t0;
  float*       qrow = outQ + ((size_t)b * DIM) * T_LEN + t0;
  for (int i = tid; i < TTILE * DIM; i += NTHREADS) {
    const int t = i & (TTILE - 1);
    const int d = i >> 7;
    const int code = idxArr[t];
    const float q = E[(size_t)code * DIM + d];
    const float x = xrow[(size_t)d * T_LEN + t];
    const float df = q - x;
    lsum += df * df;
    qrow[(size_t)d * T_LEN + t] = q;     // 128 consecutive floats per d -> coalesced
  }
#pragma unroll
  for (int off = 16; off >= 1; off >>= 1) lsum += __shfl_xor(lsum, off, 32);
  if (lane == 0) redBuf[wave] = lsum;
  __syncthreads();
  if (tid == 0) {
    float tot = 0.f;
#pragma unroll
    for (int i = 0; i < 8; ++i) tot += redBuf[i];
    // loss = q_latent + 0.25*e_latent = 1.25 * mean((q-x)^2)
    atomicAdd(outLoss, tot * (1.25f / ((float)B_SZ * (float)T_LEN * (float)DIM)));
  }
  if (tid < TTILE) outIdx[(size_t)b * T_LEN + t0 + tid] = (float)idxArr[tid];
}

extern "C" void kernel_launch(void* const* d_in, const int* in_sizes, int n_in,
                              void* d_out, int out_size, void* d_ws, size_t ws_size,
                              hipStream_t stream) {
  const float* X = (const float*)d_in[0];   // [32, 256, 2048]
  const float* E = (const float*)d_in[1];   // [1024, 256]
  float* outQ    = (float*)d_out;                                   // 16,777,216
  float* outLoss = outQ + (size_t)B_SZ * DIM * T_LEN;               // 1
  float* outIdx  = outLoss + 1;                                     // 65,536

  // workspace: enorms f32[1024] | Ehi bf16[1024*256] | Elo bf16[1024*256]  (~1.05 MB)
  float*  enorms = (float*)d_ws;
  __bf16* ehi    = (__bf16*)(enorms + K_CODES);
  __bf16* elo    = ehi + (size_t)K_CODES * DIM;

  vq_prep<<<K_CODES, 256, 0, stream>>>(E, enorms, ehi, elo, outLoss);
  vq_main<<<B_SZ * (T_LEN / TTILE), NTHREADS, 0, stream>>>(X, ehi, elo, E, enorms,
                                                           outQ, outLoss, outIdx);
}
